// Dilated_10136122818699
// MI455X (gfx1250) — compile-verified, matrix operand order checked by reading, not run on hardware
//
#include <hip/hip_runtime.h>
#include <hip/hip_bf16.h>

// ---------------------------------------------------------------------------
// MI455X (gfx1250): all convs as implicit-GEMM on V_WMMA_F32_16X16X32_BF16.
// K ordered kk-major (K = kk*Cpad + ci), Cpad = round32(Cin): every fragment is
// two contiguous 16-byte runs per lane. Weights staged bf16 [Mpad][ks][Cpad]
// and double-buffered into LDS per kk via GLOBAL_LOAD_ASYNC_TO_LDS_B128
// (ASYNCcnt + s_wait_asynccnt), shared by all 4 waves of the block.
// Activations channel-interleaved [B][l][ci] bf16; 2 N-tiles per wave.
// ---------------------------------------------------------------------------

typedef __attribute__((ext_vector_type(16))) __bf16 v16bf;
typedef __attribute__((ext_vector_type(8)))  __bf16 v8bf;
typedef __attribute__((ext_vector_type(8)))  float  v8f;

union U16 { v16bf v; v8bf h[2]; };

#define WPB 4               // waves per block (wave32 -> 128 threads)
#define TPB (WPB * 32)
#define CPAD_MAX 352        // round32(349), largest channel count in the net

#if defined(__AMDGCN__) && __has_builtin(__builtin_amdgcn_global_load_async_to_lds_b128)
#define ASYNC_LDS 1
typedef int v4i_ __attribute__((vector_size(16)));            // b128 payload type
typedef __attribute__((address_space(1))) v4i_* as1_v4i;      // global src
typedef __attribute__((address_space(3))) v4i_* as3_v4i;      // LDS dst
#else
#define ASYNC_LDS 0
#endif

__device__ __forceinline__ void wait_async_lds() {
#if ASYNC_LDS
#if __has_builtin(__builtin_amdgcn_s_wait_asynccnt)
  __builtin_amdgcn_s_wait_asynccnt(0);
#else
  asm volatile("s_wait_asynccnt 0" ::: "memory");
#endif
#endif
}

__device__ __forceinline__ float gelu_exact(float x) {
  return 0.5f * x * (1.0f + erff(x * 0.70710678118654752440f));
}

__device__ __forceinline__ v8bf zero8() {
  v8bf z;
#pragma unroll
  for (int e = 0; e < 8; ++e) z[e] = (__bf16)0.0f;
  return z;
}

// ---- GELU: strided f32 [B][Cin][inStride] -> interleaved bf16 [B][Lin][Cpad]
__global__ void k_gelu_ilv(const float* __restrict__ in, __bf16* __restrict__ out,
                           int Cin, int Cpad, int Lin, int inStride, int inOffset) {
  long long i = (long long)blockIdx.x * blockDim.x + threadIdx.x;
  int ci = (int)(i % Cpad);
  long long q = i / Cpad;
  int l = (int)(q % Lin);
  int b = (int)(q / Lin);
  if (b >= 4) return;
  float v = 0.f;
  if (ci < Cin)
    v = gelu_exact(in[((long long)b * Cin + ci) * inStride + inOffset + l]);
  out[i] = (__bf16)v;
}

// ---- weight cast/transpose: f32 [Cout][Cin][ks] -> bf16 [Mpad][ks][Cpad] ---
// zero-padded in both M and C; optionally gathered per batch via widx (grid.y=B)
__global__ void k_wcast(const float* __restrict__ w, const int* __restrict__ widx,
                        long long wPerBatchIn, __bf16* __restrict__ out,
                        long long outPerBatch,
                        int Cout, int Cin, int ks, int Cpad, int Mpad) {
  long long per = (long long)Mpad * ks * Cpad;
  long long i = (long long)blockIdx.x * blockDim.x + threadIdx.x;
  if (i >= per) return;
  int b = blockIdx.y;
  int ci = (int)(i % Cpad);
  long long q = i / Cpad;
  int kk = (int)(q % ks);
  int m  = (int)(q / ks);
  float v = 0.f;
  if (m < Cout && ci < Cin) {
    const float* wb = w + (widx ? (long long)widx[b] * wPerBatchIn : 0);
    v = wb[((long long)m * Cin + ci) * ks + kk];
  }
  out[(long long)b * outPerBatch + i] = (__bf16)v;
}

// ---- implicit-GEMM conv on WMMA: y = BN(conv(g)+bias) (+residual) ----------
__global__ void k_conv_wmma(
    const __bf16* __restrict__ gT,   // [B][Lin][Cpad] bf16
    const __bf16* __restrict__ wst,  // [(B)][Mpad][ks][Cpad] bf16 (zero-padded)
    long long wPerBatch,
    const float* __restrict__ bias, const int* __restrict__ bidx, int biasPerBatch,
    const float* __restrict__ gamma, const float* __restrict__ beta,
    const float* __restrict__ mean,  const float* __restrict__ var,
    const float* __restrict__ residual,
    float* __restrict__ out,          // [B][Cout][Lout] f32
    int Cpad, int Cout, int Lin, int Lout, int ks, int dil, int pad)
{
  const int b     = blockIdx.z;
  const int mBase = blockIdx.y * 16;
  const int lane  = threadIdx.x & 31;
  const int half  = lane >> 4;
  const int r     = lane & 15;
  const int wave  = threadIdx.x >> 5;
  const int n0    = (blockIdx.x * WPB + wave) * 32;   // 2 N-tiles per wave
  // NOTE: no early-exit -- every wave must join LDS staging + barriers.

  const __bf16* wb = wst + (long long)b * wPerBatch;
  const __bf16* gb = gT + (long long)b * Lin * Cpad;

  // double-buffered weight slab: 16 rows x Cpad bf16 per kk
  __shared__ __align__(16) __bf16 lw[2][16 * CPAD_MAX];

  // cooperative stage of slab (mBase rows, taps kk) into buffer p
  auto stage = [&](int kk, int p) {
    const int bytes = 16 * Cpad * 2;
    for (int off = threadIdx.x * 16; off < bytes; off += TPB * 16) {
      int row = off / (Cpad * 2);
      int col = (off - row * (Cpad * 2)) >> 1;            // element column
      const __bf16* gsrc = wb + ((long long)(mBase + row) * ks + kk) * Cpad + col;
      __bf16* ldst = &lw[p][row * Cpad + col];
#if ASYNC_LDS
      __builtin_amdgcn_global_load_async_to_lds_b128((as1_v4i)gsrc, (as3_v4i)ldst,
                                                     0, 0);
#else
      *(v8bf*)ldst = *(const v8bf*)gsrc;                  // sync fallback
#endif
    }
  };

  v8f acc0 = {0.f,0.f,0.f,0.f,0.f,0.f,0.f,0.f};
  v8f acc1 = {0.f,0.f,0.f,0.f,0.f,0.f,0.f,0.f};
  const v8bf zv = zero8();
  const int eo = half * 8;     // 16-byte sub-run select per lane half

  stage(0, 0);
  wait_async_lds();
  __syncthreads();

  for (int kk = 0; kk < ks; ++kk) {
    const int cur = kk & 1;
    if (kk + 1 < ks) stage(kk + 1, cur ^ 1);   // prefetch next slab (async)

    const int sh  = kk * dil - pad;
    const int li0 = n0 + r + sh;
    const int li1 = li0 + 16;
    const bool ok0 = (unsigned)li0 < (unsigned)Lin;
    const bool ok1 = (unsigned)li1 < (unsigned)Lin;
    const __bf16* g0 = gb + (long long)li0 * Cpad;
    const __bf16* g1 = gb + (long long)li1 * Cpad;
    const __bf16* la = &lw[cur][r * Cpad];

    for (int cb = 0; cb < Cpad; cb += 32) {
      U16 A, B0, B1;
      A.h[0]  = *(const v8bf*)(la + cb + eo);             // ds_load_b128
      A.h[1]  = *(const v8bf*)(la + cb + 16 + eo);
      B0.h[0] = ok0 ? *(const v8bf*)(g0 + cb + eo)      : zv;
      B0.h[1] = ok0 ? *(const v8bf*)(g0 + cb + 16 + eo) : zv;
      B1.h[0] = ok1 ? *(const v8bf*)(g1 + cb + eo)      : zv;
      B1.h[1] = ok1 ? *(const v8bf*)(g1 + cb + 16 + eo) : zv;
      acc0 = __builtin_amdgcn_wmma_f32_16x16x32_bf16(false, A.v, false, B0.v,
                                                     (short)0, acc0, false, false);
      acc1 = __builtin_amdgcn_wmma_f32_16x16x32_bf16(false, A.v, false, B1.v,
                                                     (short)0, acc1, false, false);
    }

    if (kk + 1 < ks) {
      wait_async_lds();      // prefetch landed
      __syncthreads();       // all waves done with 'cur', next slab visible
    }
  }

  // Epilogue: D layout -> VGPR j = row (j + 8*half), lane r = column.
  const int bb = bidx ? bidx[b] * biasPerBatch : 0;
#pragma unroll
  for (int j = 0; j < 8; ++j) {
    int m = mBase + j + 8 * half;
    if (m >= Cout) continue;
    float add = bias ? bias[bb + m] : 0.f;
    float scl = 1.f, sft = 0.f;
    if (gamma) {
      scl = gamma[m] * rsqrtf(var[m] + 1e-5f);
      sft = beta[m] - mean[m] * scl;
    }
    long long rowo = ((long long)b * Cout + m) * Lout;
    int n = n0 + r;
    if (n < Lout) {
      float v = (acc0[j] + add) * scl + sft;
      if (residual) v += residual[rowo + n];
      out[rowo + n] = v;
    }
    n += 16;
    if (n < Lout) {
      float v = (acc1[j] + add) * scl + sft;
      if (residual) v += residual[rowo + n];
      out[rowo + n] = v;
    }
  }
}

// ---- z[b,l] = b2 + sum_c relu(Xc*Xa) * w2[c] -------------------------------
__global__ void k_pair(const float* __restrict__ Xc, const float* __restrict__ Xa,
                       const float* __restrict__ w2, const float* __restrict__ b2,
                       float* __restrict__ z, int C, int L) {
  int b = blockIdx.y;
  int l = blockIdx.x * blockDim.x + threadIdx.x;
  if (l >= L) return;
  const float* pc = Xc + (long long)b * C * L + l;
  const float* pa = Xa + (long long)b * C * L + l;
  float s = 0.f;
  for (int c = 0; c < C; ++c) {
    float v = pc[(long long)c * L] * pa[(long long)c * L];
    s += fmaxf(v, 0.f) * w2[c];
  }
  z[(long long)b * L + l] = s + b2[0];
}

// ---- row-wise log_softmax (one block per batch row, blockDim = 256) --------
__global__ void k_logsoftmax(const float* __restrict__ z, float* __restrict__ out, int L) {
  int b = blockIdx.x;
  const float* zb = z + (long long)b * L;
  __shared__ float sm[256];
  float m = -3.402823466e+38f;
  for (int l = threadIdx.x; l < L; l += 256) m = fmaxf(m, zb[l]);
  sm[threadIdx.x] = m; __syncthreads();
  for (int s = 128; s > 0; s >>= 1) {
    if (threadIdx.x < s) sm[threadIdx.x] = fmaxf(sm[threadIdx.x], sm[threadIdx.x + s]);
    __syncthreads();
  }
  m = sm[0]; __syncthreads();
  float acc = 0.f;
  for (int l = threadIdx.x; l < L; l += 256) acc += expf(zb[l] - m);
  sm[threadIdx.x] = acc; __syncthreads();
  for (int s = 128; s > 0; s >>= 1) {
    if (threadIdx.x < s) sm[threadIdx.x] += sm[threadIdx.x + s];
    __syncthreads();
  }
  float lse = logf(sm[0]) + m;
  for (int l = threadIdx.x; l < L; l += 256) out[(long long)b * L + l] = zb[l] - lse;
}

// ---------------------------------------------------------------------------
// Host side
// ---------------------------------------------------------------------------
struct CB { const float *b, *beta, *gamma, *mean, *var, *w; };

static inline CB cb_at(void* const* d_in, int base) {
  CB c;
  c.b     = (const float*)d_in[base + 0];
  c.beta  = (const float*)d_in[base + 1];
  c.gamma = (const float*)d_in[base + 2];
  c.mean  = (const float*)d_in[base + 3];
  c.var   = (const float*)d_in[base + 4];
  c.w     = (const float*)d_in[base + 5];
  return c;
}

extern "C" void kernel_launch(void* const* d_in, const int* in_sizes, int n_in,
                              void* d_out, int out_size, void* d_ws, size_t ws_size,
                              hipStream_t stream) {
  (void)in_sizes; (void)out_size; (void)ws_size;
  const int Bb = 4, L = 4096, NF = 128, TRIM = 1024, KT = 75, LT = 2048, LO = 1974;

  // ---- input flattening (X, then params by sorted key, then idx arrays) ----
  const float* X = (const float*)d_in[0];
  const float* assay_b    = (const float*)d_in[1];
  const float* assay_w    = (const float*)d_in[2];
  const float* celltype_b = (const float*)d_in[3];
  const float* celltype_w = (const float*)d_in[4];
  CB conv_cb = cb_at(d_in, 5);
  const float* conv2_b = (const float*)d_in[11];
  const float* conv2_w = (const float*)d_in[12];
  CB stem_conv  = cb_at(d_in, 109);
  CB stem_rconv = cb_at(d_in, 115);
  const int* ct  = (const int*)d_in[n_in - 2];
  const int* asy = (const int*)d_in[n_in - 1];

  // ---- workspace carve-up (aggressive reuse, ~92 MB) ----
  const size_t MB = 1024ull * 1024ull;
  char* base = (char*)d_ws;
  float*  xA   = (float*)(base + 0 * MB);    // 23 MB f32 ping  (also Xc later)
  float*  xB   = (float*)(base + 23 * MB);   // 23 MB f32 pong  (also Xa later)
  float*  t    = (float*)(base + 46 * MB);   // 9 MB f32 temp   (also logits z)
  __bf16* gT   = (__bf16*)(base + 55 * MB);  // 12 MB interleaved bf16 activations
  __bf16* ws_w = (__bf16*)(base + 67 * MB);  // 20 MB bf16 weight staging (reused)
  __bf16* hb   = (__bf16*)(base + 87 * MB);  // 5 MB tail input [B][2048][256]

  auto pad32 = [](int c) { return (c + 31) & ~31; };
  auto pad16 = [](int c) { return (c + 15) & ~15; };

  auto run_gelu = [&](const float* in, __bf16* outp, int Cin, int Lin,
                      int inStride, int inOffset) {
    int Cpad = pad32(Cin);
    long long total = (long long)Bb * Lin * Cpad;
    int blocks = (int)((total + 255) / 256);
    k_gelu_ilv<<<blocks, 256, 0, stream>>>(in, outp, Cin, Cpad, Lin, inStride, inOffset);
  };
  auto stage_w = [&](const float* w, const int* widx, long long wPerBatchIn,
                     __bf16* outp, int Cout, int Cin, int ks) {
    int Cpad = pad32(Cin), Mpad = pad16(Cout);
    long long per = (long long)Mpad * ks * Cpad;
    dim3 grid((unsigned)((per + 255) / 256), widx ? Bb : 1);
    k_wcast<<<grid, 256, 0, stream>>>(w, widx, wPerBatchIn, outp, per,
                                      Cout, Cin, ks, Cpad, Mpad);
  };
  auto run_conv = [&](const __bf16* gin, CB p, const float* res, float* outp,
                      int Cin, int Cout, int Lin, int Lout, int ks, int dil, int pad) {
    stage_w(p.w, nullptr, 0, ws_w, Cout, Cin, ks);
    int Cpad = pad32(Cin);
    dim3 grid((Lout + 32 * WPB - 1) / (32 * WPB), (Cout + 15) / 16, Bb);
    k_conv_wmma<<<grid, TPB, 0, stream>>>(gin, ws_w, 0, p.b, nullptr, 0,
                                          p.gamma, p.beta, p.mean, p.var,
                                          res, outp, Cpad, Cout, Lin, Lout, ks, dil, pad);
  };

  // ---- stem ----
  run_gelu(X, gT, 4, L, L, 0);
  run_conv(gT, stem_conv, nullptr, xA, 4, NF / 2, L, L, 21, 1, 10);
  run_gelu(xA, gT, NF / 2, L, L, 0);
  run_conv(gT, stem_rconv, xA, xB, NF / 2, NF / 2, L, L, 1, 1, 0);
  float* cur = xB; float* oth = xA;

  // ---- tower ----
  const int chans[7] = {64, 156, 191, 233, 285, 349, 128};
  for (int i = 0; i < 6; ++i) {
    CB c  = cb_at(d_in, 121 + 12 * i);
    CB rc = cb_at(d_in, 127 + 12 * i);
    run_gelu(cur, gT, chans[i], L, L, 0);
    run_conv(gT, c, nullptr, oth, chans[i], chans[i + 1], L, L, 5, 1, 2);
    { float* tmp = cur; cur = oth; oth = tmp; }
    run_gelu(cur, gT, chans[i + 1], L, L, 0);
    run_conv(gT, rc, cur, oth, chans[i + 1], chans[i + 1], L, L, 1, 1, 0);
    { float* tmp = cur; cur = oth; oth = tmp; }
  }

  // ---- dilated residual blocks ----
  const int DI[8] = {1, 1, 2, 3, 5, 7, 11, 17};
  for (int i = 0; i < 8; ++i) {
    CB c1 = cb_at(d_in, 13 + 12 * i);
    CB c2 = cb_at(d_in, 19 + 12 * i);
    run_gelu(cur, gT, NF, L, L, 0);
    run_conv(gT, c1, nullptr, t, NF, NF, L, L, 3, DI[i], DI[i]);
    run_gelu(t, gT, NF, L, L, 0);
    run_conv(gT, c2, cur, oth, NF, NF, L, L, 1, 1, 0);
    { float* tmp = cur; cur = oth; oth = tmp; }
  }

  // ---- trim + final 1x1 (128 -> 256) + GELU -> h ----
  run_gelu(cur, gT, NF, LT, L, TRIM);
  run_conv(gT, conv_cb, nullptr, t, NF, 2 * NF, LT, LT, 1, 1, 0);
  run_gelu(t, hb, 2 * NF, LT, LT, 0);   // hb: [B][2048][256] bf16

  // ---- per-sample tail convs (dominant GEMM: K = 75*256) ----
  {
    const long long wPerIn  = (long long)NF * 2 * NF * KT;   // f32 elems per bank entry
    const long long wPerOut = (long long)NF * KT * 256;      // staged bf16 per batch
    dim3 grid((LO + 32 * WPB - 1) / (32 * WPB), NF / 16, Bb);
    float* Xc = xA; float* Xa = xB; float* z = t;

    stage_w(celltype_w, ct, wPerIn, ws_w, NF, 2 * NF, KT);
    k_conv_wmma<<<grid, TPB, 0, stream>>>(hb, ws_w, wPerOut, celltype_b, ct, NF,
        nullptr, nullptr, nullptr, nullptr, nullptr, Xc,
        256, NF, LT, LO, KT, 1, 0);

    stage_w(assay_w, asy, wPerIn, ws_w, NF, 2 * NF, KT);
    k_conv_wmma<<<grid, TPB, 0, stream>>>(hb, ws_w, wPerOut, assay_b, asy, NF,
        nullptr, nullptr, nullptr, nullptr, nullptr, Xa,
        256, NF, LT, LO, KT, 1, 0);

    dim3 pgrid((LO + 255) / 256, Bb);
    k_pair<<<pgrid, 256, 0, stream>>>(Xc, Xa, conv2_w, conv2_b, z, NF, LO);
    k_logsoftmax<<<Bb, 256, 0, stream>>>(z, (float*)d_out, LO);
  }
}
